// Stage2TemporalModel_89945205113547
// MI455X (gfx1250) — matrix operand order repaired and checked by reading, not compiled, not run on hardware
//
#include <hip/hip_runtime.h>
#include <hip/hip_bf16.h>

typedef __attribute__((ext_vector_type(16))) __bf16 v16bf;
typedef __attribute__((ext_vector_type(8)))  float  v8f;
typedef unsigned short u16;

#define EPSF 1e-5f

// ---------- small helpers ----------
__device__ __forceinline__ float gelu_exact(float x) {
    return 0.5f * x * (1.0f + erff(x * 0.70710678118654752440f));
}

__device__ __forceinline__ u16 f2bf(float f) {
    unsigned u = __float_as_uint(f);
    unsigned r = (u + 0x7FFFu + ((u >> 16) & 1u)) >> 16;  // round-to-nearest-even
    return (u16)r;
}

struct Q2 { uint4 a; uint4 b; };
__device__ __forceinline__ v16bf make_frag(uint4 a, uint4 b) {
    Q2 q; q.a = a; q.b = b;
    return __builtin_bit_cast(v16bf, q);
}

// ================= k0: weight conversion fp32 -> bf16 =================
__global__ __launch_bounds__(256) void k0_convert(
    const float* __restrict__ wmix, const float* __restrict__ edw1,
    const float* __restrict__ esw1,
    u16* __restrict__ Wb, u16* __restrict__ w1ed, u16* __restrict__ w1es)
{
    int i = blockIdx.x * 256 + threadIdx.x;
    const int NW = 512 * 1536;          // 786432
    const int NH = 256 * 512;           // 131072
    if (i < NW) {
        Wb[i] = f2bf(wmix[i]);
    } else if (i < NW + NH) {
        w1ed[i - NW] = f2bf(edw1[i - NW]);
    } else if (i < NW + 2 * NH) {
        w1es[i - NW - NH] = f2bf(esw1[i - NW - NH]);
    }
}

// ================= k1: interp + dwconv3/5 + BN + GELU =================
// thread id -> (b = id>>9, c = id&511). Writes bf16 cat rows X[bt,1536]
// and fp32 tf_t[bt,512].
__global__ __launch_bounds__(256) void k1_prep(
    const float* __restrict__ x, const float* __restrict__ w3,
    const float* __restrict__ bn3g, const float* __restrict__ bn3b,
    const float* __restrict__ bn3m, const float* __restrict__ bn3v,
    const float* __restrict__ w5,
    const float* __restrict__ bn5g, const float* __restrict__ bn5b,
    const float* __restrict__ bn5m, const float* __restrict__ bn5v,
    u16* __restrict__ Xb, float* __restrict__ tf32)
{
    int id = blockIdx.x * 256 + threadIdx.x;     // 0 .. B*C-1
    int b = id >> 9, c = id & 511;

    const float4* x4 = (const float4*)(x + (size_t)id * 64);
    float tf[32];
#pragma unroll
    for (int i = 0; i < 16; ++i) {
        float4 q = x4[i];
        tf[2 * i]     = 0.5f * (q.x + q.y);
        tf[2 * i + 1] = 0.5f * (q.z + q.w);
    }

    float k30 = w3[c * 3 + 0], k31 = w3[c * 3 + 1], k32 = w3[c * 3 + 2];
    float i3 = bn3g[c] * rsqrtf(bn3v[c] + EPSF), m3 = bn3m[c], a3 = bn3b[c];
    float k5w[5];
#pragma unroll
    for (int k = 0; k < 5; ++k) k5w[k] = w5[c * 5 + k];
    float i5 = bn5g[c] * rsqrtf(bn5v[c] + EPSF), m5 = bn5m[c], a5 = bn5b[c];

    size_t rowbase = (size_t)b * 32;
#pragma unroll
    for (int t = 0; t < 32; ++t) {
        float tm1 = (t > 0)  ? tf[t - 1] : 0.f;
        float tp1 = (t < 31) ? tf[t + 1] : 0.f;
        float tm2 = (t > 1)  ? tf[t - 2] : 0.f;
        float tp2 = (t < 30) ? tf[t + 2] : 0.f;
        float c3 = k30 * tm1 + k31 * tf[t] + k32 * tp1;
        float c5 = k5w[0] * tm2 + k5w[1] * tm1 + k5w[2] * tf[t] + k5w[3] * tp1 + k5w[4] * tp2;
        float g3 = gelu_exact((c3 - m3) * i3 + a3);
        float g5 = gelu_exact((c5 - m5) * i5 + a5);
        size_t ro = (rowbase + t) * 1536;
        Xb[ro + c]        = f2bf(tf[t]);
        Xb[ro + 512 + c]  = f2bf(g3);
        Xb[ro + 1024 + c] = f2bf(g5);
        tf32[(rowbase + t) * 512 + c] = tf[t];
    }
}

// ================= k2: mix GEMM + BN + GELU + residual + LayerNorm =================
// Block: 256 threads (8 waves). Tile: 32 bt-rows (one full sample) x 512 out
// channels. A [32 x 1536] bf16 streamed through LDS in K-chunks of 128
// (async to LDS, double buffered). Each wave owns 4 N-tiles of 16 (64 out
// channels) x 2 M-tiles; B fragments are register-pipelined one k-step ahead
// and reused across both M-tiles (halves L2 traffic on wmix vs a 16-row tile).
#define KC 128
__global__ __launch_bounds__(256) void k2_mix_ln(
    const u16* __restrict__ Xb, const float* __restrict__ tf32,
    const u16* __restrict__ Wb, const float* __restrict__ pos,
    const float* __restrict__ bnmg, const float* __restrict__ bnmb,
    const float* __restrict__ bnmm, const float* __restrict__ bnmv,
    const float* __restrict__ lng, const float* __restrict__ lnb,
    float* __restrict__ feats)
{
    __shared__ __align__(16) u16 As[2][32 * KC];
    __shared__ float fpbuf[16 * 512];
    __shared__ float muL[16], rsL[16];

    const int tid  = threadIdx.x;
    const int wave = tid >> 5;
    const int lane = tid & 31;
    const int r0 = blockIdx.x * 32;   // global bt row base (t0 == 0)

    // async-copy one 32x128 bf16 A chunk into LDS (each thread moves 2x16B)
    auto issue = [&](int buf, int kc) {
#pragma unroll
        for (int i = 0; i < 2; ++i) {
            int seg  = tid + i * 256;
            int row  = seg >> 4;   // 0..31
            int segc = seg & 15;   // 16 segments x 8 bf16
            unsigned lds_off = (unsigned)(unsigned long long)(&As[buf][row * KC + segc * 8]);
            unsigned goff = (unsigned)(((r0 + row) * 1536 + kc + segc * 8) * 2);
            asm volatile("global_load_async_to_lds_b128 %0, %1, %2"
                         :: "v"(lds_off), "v"(goff), "s"(Xb) : "memory");
        }
    };

    const int ncol  = lane & 15;          // N within tile
    const int base0 = (lane >> 4) * 8;    // A K base (ISA A-matrix layout)
    const int khi   = (lane >> 4) * 8;    // B K base (same order as A)

    // load the wave's 4 B fragments for global k-step `kglob`
    auto loadB = [&](uint4 (&dst)[4][2], int kglob) {
#pragma unroll
        for (int j = 0; j < 4; ++j) {
            const u16* bp = Wb + (wave * 64 + j * 16 + ncol) * 1536 + kglob + khi;
            dst[j][0] = *(const uint4*)bp;
            dst[j][1] = *(const uint4*)(bp + 16);
        }
    };

    v8f acc[2][4] = {};        // [jm][jn]
    uint4 breg[4][2];
    issue(0, 0);
    loadB(breg, 0);
    const int NCHUNK = 1536 / KC;   // 12

    for (int ci = 0; ci < NCHUNK; ++ci) {
        int buf = ci & 1;
        if (ci + 1 < NCHUNK) {
            issue(buf ^ 1, (ci + 1) * KC);
            asm volatile("s_wait_asynccnt 0x2" ::: "memory");   // chunk ci resident
        } else {
            asm volatile("s_wait_asynccnt 0x0" ::: "memory");
        }
        __syncthreads();

#pragma unroll
        for (int kk = 0; kk < KC; kk += 32) {
            int kglob = ci * KC + kk;
            // prefetch next k-step's B fragments (hides L2 latency under WMMAs)
            uint4 bnew[4][2];
            if (kglob + 32 < 1536) loadB(bnew, kglob + 32);

            v16bf af[2];
#pragma unroll
            for (int jm = 0; jm < 2; ++jm) {
                const u16* ar = &As[buf][(jm * 16 + ncol) * KC + kk];
                uint4 a0 = *(const uint4*)(ar + base0);
                uint4 a1 = *(const uint4*)(ar + base0 + 16);
                af[jm] = make_frag(a0, a1);
            }
#pragma unroll
            for (int jn = 0; jn < 4; ++jn) {
                v16bf bf = make_frag(breg[jn][0], breg[jn][1]);
#pragma unroll
                for (int jm = 0; jm < 2; ++jm)
                    acc[jm][jn] = __builtin_amdgcn_wmma_f32_16x16x32_bf16(
                        false, af[jm], false, bf, (short)0, acc[jm][jn], false, false);
            }
            if (kglob + 32 < 1536) {
#pragma unroll
                for (int j = 0; j < 4; ++j) {
                    breg[j][0] = bnew[j][0];
                    breg[j][1] = bnew[j][1];
                }
            }
        }
        __syncthreads();   // protect LDS buffer before it is overwritten
    }

    // epilogue: two passes of 16 rows through fpbuf (BN+GELU, residual,
    // pos_embed, then cross-wave LayerNorm)
    const int mbase = (lane >> 4) * 8;
#pragma unroll
    for (int pass = 0; pass < 2; ++pass) {
#pragma unroll
        for (int jn = 0; jn < 4; ++jn) {
            int o = wave * 64 + jn * 16 + ncol;
            float inv = bnmg[o] * rsqrtf(bnmv[o] + EPSF);
            float mm = bnmm[o], ab = bnmb[o];
#pragma unroll
            for (int r = 0; r < 8; ++r) {
                int M = mbase + r;            // row within pass (0..15)
                int t = pass * 16 + M;        // t within sample
                float ctx = gelu_exact((acc[pass][jn][r] - mm) * inv + ab);
                float fp = tf32[(size_t)(r0 + t) * 512 + o] + ctx + pos[t * 512 + o];
                fpbuf[M * 512 + o] = fp;
            }
        }
        __syncthreads();

        // per-row LayerNorm stats (16 rows x 16 threads, 32 elems each)
        {
            int row = tid >> 4, seg = tid & 15;
            float s1 = 0.f, s2 = 0.f;
#pragma unroll
            for (int i = 0; i < 32; ++i) {
                float fv = fpbuf[row * 512 + seg * 32 + i];
                s1 += fv; s2 += fv * fv;
            }
#pragma unroll
            for (int w = 8; w >= 1; w >>= 1) {
                s1 += __shfl_xor(s1, w, 16);
                s2 += __shfl_xor(s2, w, 16);
            }
            if (seg == 0) {
                float mu = s1 * (1.f / 512.f);
                float var = s2 * (1.f / 512.f) - mu * mu;
                muL[row] = mu;
                rsL[row] = rsqrtf(var + EPSF);
            }
        }
        __syncthreads();

        // normalize + affine, write feats
        {
            int row = tid >> 4, seg = tid & 15;
            float mu = muL[row], rs = rsL[row];
#pragma unroll
            for (int i = 0; i < 32; ++i) {
                int o = seg * 32 + i;
                float fv = (fpbuf[row * 512 + o] - mu) * rs * lng[o] + lnb[o];
                feats[(size_t)(r0 + pass * 16 + row) * 512 + o] = fv;
            }
        }
        __syncthreads();   // fpbuf reused by next pass
    }
}

// ================= k3: attention MLPs + softmax + pooling =================
// One block per b. 256 threads (8 waves).
__global__ __launch_bounds__(256) void k3_attn_pool(
    const float* __restrict__ feats,
    const u16* __restrict__ w1ed, const u16* __restrict__ w1es,
    const float* __restrict__ edlng, const float* __restrict__ edlnb,
    const float* __restrict__ edb1,  const float* __restrict__ edw2,
    const float* __restrict__ edb2,
    const float* __restrict__ eslng, const float* __restrict__ eslnb,
    const float* __restrict__ esb1,  const float* __restrict__ esw2,
    const float* __restrict__ esb2,
    const float* __restrict__ tped,  const float* __restrict__ tpes,
    float* __restrict__ pooled, float* __restrict__ pair)
{
    __shared__ __align__(16) u16 Ab[32 * 512];
    __shared__ float muL[32], rsL[32];
    __shared__ float sL[32];
    __shared__ float awL[2][32];

    const int tid = threadIdx.x;
    const int wave = tid >> 5;
    const int lane = tid & 31;
    const int b = blockIdx.x;
    const float* F = feats + (size_t)b * 32 * 512;

    // per-row mean / rstd of feats (shared by both attention LNs)
    {
        int row = tid >> 3, seg = tid & 7;
        float s1 = 0.f, s2 = 0.f;
        for (int i = 0; i < 64; ++i) {
            float fv = F[row * 512 + seg * 64 + i];
            s1 += fv; s2 += fv * fv;
        }
#pragma unroll
        for (int w = 4; w >= 1; w >>= 1) {
            s1 += __shfl_xor(s1, w, 8);
            s2 += __shfl_xor(s2, w, 8);
        }
        if (seg == 0) {
            float mu = s1 * (1.f / 512.f);
            float var = s2 * (1.f / 512.f) - mu * mu;
            muL[row] = mu;
            rsL[row] = rsqrtf(var + EPSF);
        }
    }
    __syncthreads();

    const int ncol  = lane & 15;
    const int khi   = (lane >> 4) * 8;
    const int base0 = (lane >> 4) * 8;

    for (int p = 0; p < 2; ++p) {
        const u16* W1 = p ? w1es : w1ed;
        const float* lg = p ? eslng : edlng;
        const float* lb = p ? eslnb : edlnb;
        const float* b1 = p ? esb1  : edb1;
        const float* w2 = p ? esw2  : edw2;
        float b2   = p ? esb2[0] : edb2[0];
        float temp = p ? tpes[0] : tped[0];
        temp = fminf(fmaxf(temp, 0.25f), 4.0f);

        // build LN-scaled bf16 activations
        {
            int row = tid >> 3, c0 = (tid & 7) * 64;
            float mu = muL[row], rs = rsL[row];
            for (int i = 0; i < 64; ++i) {
                int c = c0 + i;
                float nv = (F[row * 512 + c] - mu) * rs * lg[c] + lb[c];
                Ab[row * 512 + c] = f2bf(nv);
            }
        }
        if (tid < 32) sL[tid] = 0.f;
        __syncthreads();

        // GEMM [32 x 512] @ [512 -> 256]; wave owns 32 N cols, both M tiles.
        // B fragments register-pipelined one k-step ahead.
        auto loadB3 = [&](uint4 (&dst)[2][2], int k) {
#pragma unroll
            for (int jn = 0; jn < 2; ++jn) {
                const u16* bp = W1 + (wave * 32 + jn * 16 + ncol) * 512 + k + khi;
                dst[jn][0] = *(const uint4*)bp;
                dst[jn][1] = *(const uint4*)(bp + 16);
            }
        };

        v8f acc[2][2] = {};
        uint4 bcur[2][2];
        loadB3(bcur, 0);
#pragma unroll 4
        for (int k = 0; k < 512; k += 32) {
            uint4 bnew[2][2];
            if (k + 32 < 512) loadB3(bnew, k + 32);

            v16bf af[2];
#pragma unroll
            for (int jm = 0; jm < 2; ++jm) {
                const u16* ar = &Ab[(jm * 16 + ncol) * 512];
                uint4 a0 = *(const uint4*)(ar + k + base0);
                uint4 a1 = *(const uint4*)(ar + k + base0 + 16);
                af[jm] = make_frag(a0, a1);
            }
#pragma unroll
            for (int jn = 0; jn < 2; ++jn) {
                v16bf bfr = make_frag(bcur[jn][0], bcur[jn][1]);
#pragma unroll
                for (int jm = 0; jm < 2; ++jm)
                    acc[jm][jn] = __builtin_amdgcn_wmma_f32_16x16x32_bf16(
                        false, af[jm], false, bfr, (short)0, acc[jm][jn], false, false);
            }
            if (k + 32 < 512) {
#pragma unroll
                for (int jn = 0; jn < 2; ++jn) {
                    bcur[jn][0] = bnew[jn][0];
                    bcur[jn][1] = bnew[jn][1];
                }
            }
        }

        // GELU + dot with w2, reduce per-row into sL
#pragma unroll
        for (int jn = 0; jn < 2; ++jn) {
            int n = wave * 32 + jn * 16 + ncol;
            float b1v = b1[n], w2v = w2[n];
#pragma unroll
            for (int jm = 0; jm < 2; ++jm) {
#pragma unroll
                for (int r = 0; r < 8; ++r) {
                    int row = jm * 16 + (lane >> 4) * 8 + r;
                    float h = gelu_exact(acc[jm][jn][r] + b1v);
                    atomicAdd(&sL[row], h * w2v);
                }
            }
        }
        __syncthreads();

        // softmax over T=32 (single wave)
        if (tid < 32) {
            float sv = (sL[tid] + b2) / temp;
            float mx = sv;
#pragma unroll
            for (int w = 16; w >= 1; w >>= 1) mx = fmaxf(mx, __shfl_xor(mx, w, 32));
            float e = expf(sv - mx);
            float ssum = e;
#pragma unroll
            for (int w = 16; w >= 1; w >>= 1) ssum += __shfl_xor(ssum, w, 32);
            float aw = e / ssum;
            awL[p][tid] = aw;
            pair[((size_t)b * 32 + tid) * 2 + p] = aw;
        }
        __syncthreads();
    }

    // pooled[c] = sum_t feats[t,c] * 0.5*(aw_ed[t]+aw_es[t])
    for (int c = tid; c < 512; c += 256) {
        float accp = 0.f;
#pragma unroll
        for (int t = 0; t < 32; ++t)
            accp += F[t * 512 + c] * (0.5f * (awL[0][t] + awL[1][t]));
        pooled[(size_t)b * 512 + c] = accp;
    }
}

// ================= launch =================
extern "C" void kernel_launch(void* const* d_in, const int* in_sizes, int n_in,
                              void* d_out, int out_size, void* d_ws, size_t ws_size,
                              hipStream_t stream) {
    (void)in_sizes; (void)n_in; (void)out_size; (void)ws_size;

    const float* x     = (const float*)d_in[0];
    const float* pos   = (const float*)d_in[1];
    const float* w3    = (const float*)d_in[2];
    const float* bn3g  = (const float*)d_in[3];
    const float* bn3b  = (const float*)d_in[4];
    const float* bn3m  = (const float*)d_in[5];
    const float* bn3v  = (const float*)d_in[6];
    const float* w5    = (const float*)d_in[7];
    const float* bn5g  = (const float*)d_in[8];
    const float* bn5b  = (const float*)d_in[9];
    const float* bn5m  = (const float*)d_in[10];
    const float* bn5v  = (const float*)d_in[11];
    const float* wmix  = (const float*)d_in[12];
    const float* bnmg  = (const float*)d_in[13];
    const float* bnmb  = (const float*)d_in[14];
    const float* bnmm  = (const float*)d_in[15];
    const float* bnmv  = (const float*)d_in[16];
    const float* lng   = (const float*)d_in[17];
    const float* lnb   = (const float*)d_in[18];
    const float* edlng = (const float*)d_in[19];
    const float* edlnb = (const float*)d_in[20];
    const float* edw1  = (const float*)d_in[21];
    const float* edb1  = (const float*)d_in[22];
    const float* edw2  = (const float*)d_in[23];
    const float* edb2  = (const float*)d_in[24];
    const float* eslng = (const float*)d_in[25];
    const float* eslnb = (const float*)d_in[26];
    const float* esw1  = (const float*)d_in[27];
    const float* esb1  = (const float*)d_in[28];
    const float* esw2  = (const float*)d_in[29];
    const float* esb2  = (const float*)d_in[30];
    const float* tped  = (const float*)d_in[31];
    const float* tpes  = (const float*)d_in[32];

    // outputs: feats [B,T,C], pooled [B,C], pair [B,T,2]
    float* feats  = (float*)d_out;
    float* pooled = feats + (size_t)2048 * 32 * 512;   // 33,554,432
    float* pair   = pooled + (size_t)2048 * 512;       // +1,048,576

    // workspace layout
    char* ws = (char*)d_ws;
    u16*   Xb   = (u16*)ws;                                   // 201,326,592 B
    float* tf32 = (float*)(ws + 201326592ull);                // 134,217,728 B
    u16*   Wb   = (u16*)(ws + 201326592ull + 134217728ull);   //   1,572,864 B
    u16*   w1ed = (u16*)(ws + 201326592ull + 134217728ull + 1572864ull);
    u16*   w1es = (u16*)(ws + 201326592ull + 134217728ull + 1572864ull + 262144ull);

    k0_convert<<<4096, 256, 0, stream>>>(wmix, edw1, esw1, Wb, w1ed, w1es);
    k1_prep<<<4096, 256, 0, stream>>>(x, w3, bn3g, bn3b, bn3m, bn3v,
                                      w5, bn5g, bn5b, bn5m, bn5v, Xb, tf32);
    k2_mix_ln<<<2048, 256, 0, stream>>>(Xb, tf32, Wb, pos,
                                        bnmg, bnmb, bnmm, bnmv, lng, lnb, feats);
    k3_attn_pool<<<2048, 256, 0, stream>>>(feats, w1ed, w1es,
                                           edlng, edlnb, edb1, edw2, edb2,
                                           eslng, eslnb, esb1, esw2, esb2,
                                           tped, tpes, pooled, pair);
}
